// LawinHead2_13735305413120
// MI455X (gfx1250) — matrix operand order repaired and checked by reading, not compiled, not run on hardware
//
#include <hip/hip_runtime.h>
#include <hip/hip_bf16.h>
#include <stdint.h>
#include <stddef.h>

// ---------------------------------------------------------------------------
// Types / helpers
// ---------------------------------------------------------------------------
typedef __bf16 bf16_t;
typedef bf16_t v16bf __attribute__((ext_vector_type(16)));
typedef bf16_t v8bf  __attribute__((ext_vector_type(8)));
typedef float  v8f   __attribute__((ext_vector_type(8)));

#define BN_INV 0.9999950000374996f  /* 1/sqrt(1+1e-5) */

static __device__ __forceinline__ bf16_t f2bf(float x) {
  unsigned u = __builtin_bit_cast(unsigned, x);
  u += 0x7FFFu + ((u >> 16) & 1u);               // round-to-nearest-even
  unsigned short h = (unsigned short)(u >> 16);
  return __builtin_bit_cast(bf16_t, h);
}
static __device__ __forceinline__ float bf2f(bf16_t h) {
  unsigned u = ((unsigned)__builtin_bit_cast(unsigned short, h)) << 16;
  return __builtin_bit_cast(float, u);
}

// ---------------------------------------------------------------------------
// Generic WMMA bf16 GEMM:  Out[n][m] (+epilogue) = sum_k A[m][k] * B[n][k]
//   A: weights, row-major [M][lda]   (lda multiple of 32)
//   B: activations pixel-major [N][ldb] (ldb multiple of 32) == column-major B
//   Wave tile 16(M) x 64(N), block = 8 waves -> 64(M) x 128(N)
// Fragment layouts used here (contiguous k order, same for A and B; not AMD's table order):
//   A 16x32: lanes 0-15 row M=l k:{0..15};       lanes 16-31 row M=l-16 k:{16..31}
//   B 32x16: lanes 0-15 col N=l k:{0..15};       lanes 16-31 col N=l-16 k:{16..31}
// -> both are contiguous 16B/32B loads with these layouts.
// ---------------------------------------------------------------------------
template <bool OUTF32>
__global__ __launch_bounds__(256) void k_gemm(
    const bf16_t* __restrict__ A, const bf16_t* __restrict__ B,
    void* __restrict__ Out, int ldo, int ooff,
    int M, int N, int K, int lda, int ldb,
    const float* __restrict__ scale, float smul,
    const float* __restrict__ bias,
    const bf16_t* __restrict__ resid, int ldr,
    const int* __restrict__ cmap, int relu) {
  const int lane = threadIdx.x & 31;
  const int wave = threadIdx.x >> 5;
  const int mw = wave & 3, nw = wave >> 2;
  const int tm = blockIdx.y * 64 + mw * 16;
  const int tn = blockIdx.x * 128 + nw * 64;
  if (tm >= M || tn >= N) return;              // uniform per wave -> EXEC all-1s
  const int hi = lane >> 4;
  const int l15 = lane & 15;

  int arow = tm + l15; if (arow > M - 1) arow = M - 1;
  const bf16_t* Ap = A + (size_t)arow * lda + hi * 16;

  const bf16_t* Bp[4];
#pragma unroll
  for (int j = 0; j < 4; ++j) {
    int c = tn + j * 16 + l15; if (c > N - 1) c = N - 1;
    Bp[j] = B + (size_t)c * ldb + hi * 16;
  }

  const v8f vz = {0.f, 0.f, 0.f, 0.f, 0.f, 0.f, 0.f, 0.f};
  v8f acc[4];
#pragma unroll
  for (int j = 0; j < 4; ++j) acc[j] = vz;

  for (int k0 = 0; k0 < K; k0 += 32) {
    v8bf alo = *(const v8bf*)(Ap + k0);
    v8bf ahi = *(const v8bf*)(Ap + k0 + 8);
    v16bf af;
#pragma unroll
    for (int i = 0; i < 8; ++i) { af[i] = alo[i]; af[8 + i] = ahi[i]; }
#pragma unroll
    for (int j = 0; j < 4; ++j) {
      v16bf bf = *(const v16bf*)(Bp[j] + k0);
      acc[j] = __builtin_amdgcn_wmma_f32_16x16x32_bf16(
          false, af, false, bf, (short)0, acc[j], false, false);
    }
  }

#pragma unroll
  for (int j = 0; j < 4; ++j) {
    int nloc = tn + j * 16 + l15;
    if (nloc >= N) continue;
    int nout = cmap ? cmap[nloc] : nloc;
#pragma unroll
    for (int rr = 0; rr < 8; ++rr) {
      int m = tm + rr + hi * 8;                // D layout: vgpr rr, halves -> M
      if (m >= M) continue;
      float v = acc[j][rr];
      if (scale) v *= smul * scale[m];
      if (bias)  v += bias[m];
      if (resid) v += bf2f(resid[(size_t)nloc * ldr + m]);
      if (relu && v < 0.f) v = 0.f;
      size_t oi = (size_t)nout * ldo + ooff + m;
      if (OUTF32) ((float*)Out)[oi] = v;
      else        ((bf16_t*)Out)[oi] = f2bf(v);
    }
  }
}

// ---------------------------------------------------------------------------
// Elementwise / utility kernels
// ---------------------------------------------------------------------------
__global__ void k_cvt_w(const float* __restrict__ W, bf16_t* __restrict__ A,
                        int M, int K, int Kp) {
  size_t i = (size_t)blockIdx.x * blockDim.x + threadIdx.x;
  if (i >= (size_t)M * Kp) return;
  int m = (int)(i / Kp), k = (int)(i % Kp);
  A[i] = (k < K) ? f2bf(W[(size_t)m * K + k]) : f2bf(0.f);
}

// ds_W (512,512,r,r) -> A[o][(p*r+q)*512 + c]  (k ordered (p,q,c), channel-contig)
__global__ void k_cvt_ds(const float* __restrict__ W, bf16_t* __restrict__ A, int r) {
  int rr = r * r;
  size_t total = (size_t)512 * 512 * rr;
  size_t i = (size_t)blockIdx.x * blockDim.x + threadIdx.x;
  if (i >= total) return;
  int o = (int)(i / ((size_t)512 * rr));
  int rem = (int)(i % ((size_t)512 * rr));
  int pq = rem / 512, c = rem % 512;
  A[i] = f2bf(W[((size_t)o * 512 + c) * rr + pq]);
}

// f32 [C][HW] channel-major -> bf16 [HW][Kp] pixel-major (zero pad K..Kp)
__global__ void k_transpose(const float* __restrict__ X, bf16_t* __restrict__ T,
                            int C, int HW, int Kp) {
  size_t i = (size_t)blockIdx.x * blockDim.x + threadIdx.x;
  if (i >= (size_t)HW * Kp) return;
  int p = (int)(i / Kp), k = (int)(i % Kp);
  T[i] = (k < C) ? f2bf(X[(size_t)k * HW + p]) : f2bf(0.f);
}

// Bilinear (half-pixel, edge clamp) on pixel-major channels
__global__ void k_resize(const bf16_t* __restrict__ S, int hs, int ws_, int lds_,
                         bf16_t* __restrict__ D, int hd, int wd, int ldd, int doff,
                         int C) {
  size_t i = (size_t)blockIdx.x * blockDim.x + threadIdx.x;
  if (i >= (size_t)hd * wd * C) return;
  int c = (int)(i % C);
  int p = (int)(i / C);
  int y = p / wd, x = p % wd;
  float fy = (y + 0.5f) * (float)hs / (float)hd - 0.5f;
  float fx = (x + 0.5f) * (float)ws_ / (float)wd - 0.5f;
  int y0f = (int)floorf(fy), x0f = (int)floorf(fx);
  float wy = fy - (float)y0f, wx = fx - (float)x0f;
  int y0 = min(max(y0f, 0), hs - 1), y1 = min(max(y0f + 1, 0), hs - 1);
  int x0 = min(max(x0f, 0), ws_ - 1), x1 = min(max(x0f + 1, 0), ws_ - 1);
  float v00 = bf2f(S[(size_t)(y0 * ws_ + x0) * lds_ + c]);
  float v01 = bf2f(S[(size_t)(y0 * ws_ + x1) * lds_ + c]);
  float v10 = bf2f(S[(size_t)(y1 * ws_ + x0) * lds_ + c]);
  float v11 = bf2f(S[(size_t)(y1 * ws_ + x1) * lds_ + c]);
  float v = (1.f - wy) * ((1.f - wx) * v00 + wx * v01) + wy * ((1.f - wx) * v10 + wx * v11);
  D[(size_t)p * ldd + doff + c] = f2bf(v);
}

__global__ void k_meanpool(const bf16_t* __restrict__ Cc, float* __restrict__ mean) {
  int c = blockIdx.x * blockDim.x + threadIdx.x;
  if (c >= 512) return;
  float s = 0.f;
  for (int p = 0; p < 2304; ++p) s += bf2f(Cc[(size_t)p * 512 + c]);
  mean[c] = s * (1.f / 2304.f);
}

__global__ void k_ip_broadcast(const float* __restrict__ ipW, const float* __restrict__ ipb,
                               const float* __restrict__ mean, bf16_t* __restrict__ cat,
                               int ldc) {
  int o = blockIdx.x * blockDim.x + threadIdx.x;
  if (o >= 512) return;
  float s = ipb[o];
  for (int c = 0; c < 512; ++c) s += ipW[(size_t)o * 512 + c] * mean[c];
  bf16_t h = f2bf(s);
  for (int p = 0; p < 2304; ++p) cat[(size_t)p * ldc + 512 + o] = h;
}

// deduplicated unfold gather: Bds[(s,t)][(p*r+q)*512+c] = _c[r*s+p-pad, r*t+q-pad][c]
__global__ void k_gather_ds(const bf16_t* __restrict__ Cc, bf16_t* __restrict__ Bds,
                            int r, int Ns, int pad) {
  int rr = r * r;
  size_t total = (size_t)Ns * Ns * rr * 512;
  size_t i = (size_t)blockIdx.x * blockDim.x + threadIdx.x;
  if (i >= total) return;
  int c = (int)(i % 512);
  size_t t1 = i / 512;
  int pq = (int)(t1 % rr);
  int n = (int)(t1 / rr);
  int s = n / Ns, t = n % Ns;
  int p = pq / r, q = pq % r;
  int gy = r * s + p - pad, gx = r * t + q - pad;
  float v = 0.f;
  if (gy >= 0 && gy < 48 && gx >= 0 && gx < 48)
    v = bf2f(Cc[(size_t)(gy * 48 + gx) * 512 + c]);
  Bds[i] = f2bf(v);
}

__global__ void k_layernorm(const float* __restrict__ E, bf16_t* __restrict__ O,
                            const float* __restrict__ g, const float* __restrict__ be,
                            int Nn) {
  int n = blockIdx.x;
  if (n >= Nn) return;
  __shared__ float red[256];
  const float* x = E + (size_t)n * 512;
  float s = 0.f;
  for (int c = threadIdx.x; c < 512; c += 256) s += x[c];
  red[threadIdx.x] = s; __syncthreads();
  for (int st = 128; st > 0; st >>= 1) {
    if ((int)threadIdx.x < st) red[threadIdx.x] += red[threadIdx.x + st];
    __syncthreads();
  }
  float mu = red[0] * (1.f / 512.f); __syncthreads();
  float v = 0.f;
  for (int c = threadIdx.x; c < 512; c += 256) { float d = x[c] - mu; v += d * d; }
  red[threadIdx.x] = v; __syncthreads();
  for (int st = 128; st > 0; st >>= 1) {
    if ((int)threadIdx.x < st) red[threadIdx.x] += red[threadIdx.x + st];
    __syncthreads();
  }
  float inv = rsqrtf(red[0] * (1.f / 512.f) + 1e-5f);
  for (int c = threadIdx.x; c < 512; c += 256)
    O[(size_t)n * 512 + c] = f2bf((x[c] - mu) * inv * g[c] + be[c]);
}

// patch-col (b*64+n) -> global pixel index
__global__ void k_make_cmap(int* __restrict__ cmap) {
  int cq = blockIdx.x * blockDim.x + threadIdx.x;
  if (cq >= 2304) return;
  int b = cq >> 6, n = cq & 63;
  int bi = b / 6, bj = b % 6, i = n >> 3, j = n & 7;
  cmap[cq] = (8 * bi + i) * 48 + (8 * bj + j);
}

__global__ void k_gather_rows(const bf16_t* __restrict__ S, bf16_t* __restrict__ D,
                              const int* __restrict__ cmap, int Nc, int C) {
  size_t i = (size_t)blockIdx.x * blockDim.x + threadIdx.x;
  if (i >= (size_t)Nc * C) return;
  int c = (int)(i % C); int cq = (int)(i / C);
  D[i] = S[(size_t)cmap[cq] * C + c];
}

// ctx2 = ctx + pm_W-mix + pm_b, gathered per patch from deduplicated Ectx grid
__global__ void k_mix(const bf16_t* __restrict__ Ectx, const float* __restrict__ pmW,
                      const float* __restrict__ pmb, bf16_t* __restrict__ ctx2,
                      int r, int Ns, int head) {
  size_t i = (size_t)blockIdx.x * blockDim.x + threadIdx.x;
  if (i >= (size_t)2304 * 512) return;
  int c = (int)(i % 512);
  int cq = (int)(i / 512);
  int b = cq >> 6, m = cq & 63;
  int bi = b / 6, bj = b % 6;
  int sb = (8 / r) * bi, tb = (8 / r) * bj;
  int h = c / (512 / head);
  int mi = m >> 3, mj = m & 7;
  const float* w = pmW + ((size_t)h * 64 + m) * 64;
  float self = bf2f(Ectx[(size_t)((sb + mi) * Ns + (tb + mj)) * 512 + c]);
  float s = 0.f;
  for (int n = 0; n < 64; ++n) {
    int ni = n >> 3, nj = n & 7;
    s += w[n] * bf2f(Ectx[(size_t)((sb + ni) * Ns + (tb + nj)) * 512 + c]);
  }
  ctx2[i] = f2bf(self + s + pmb[h * 64 + m]);
}

// per (patch, head) attention: softmax(th^T ph / sqrt(d)) g^T
__global__ void k_attn(const bf16_t* __restrict__ thc, const bf16_t* __restrict__ phc,
                       const bf16_t* __restrict__ gc, bf16_t* __restrict__ yc,
                       int d, int head) {
  int blk = blockIdx.x;
  int b = blk / head, h = blk % head;
  extern __shared__ float sm[];
  float* sth = sm;
  float* sph = sm + 64 * d;
  float* sg  = sm + 128 * d;
  for (int idx = threadIdx.x; idx < 64 * d; idx += 64) {
    int pos = idx / d, dd = idx % d;
    size_t col = (size_t)(b * 64 + pos) * 256 + h * d;
    sth[idx] = bf2f(thc[col + dd]);
    sph[idx] = bf2f(phc[col + dd]);
    sg[idx]  = bf2f(gc[col + dd]);
  }
  __syncthreads();
  int n = threadIdx.x;
  float logits[64];
  float mx = -1e30f;
  float scl = rsqrtf((float)d);
  for (int m = 0; m < 64; ++m) {
    float s = 0.f;
    for (int dd = 0; dd < d; ++dd) s += sth[n * d + dd] * sph[m * d + dd];
    s *= scl; logits[m] = s; mx = fmaxf(mx, s);
  }
  float den = 0.f;
  for (int m = 0; m < 64; ++m) { logits[m] = __expf(logits[m] - mx); den += logits[m]; }
  float inv = 1.f / den;
  for (int dd = 0; dd < d; ++dd) {
    float a = 0.f;
    for (int m = 0; m < 64; ++m) a += logits[m] * sg[m * d + dd];
    yc[(size_t)(b * 64 + n) * 256 + h * d + dd] = f2bf(a * inv);
  }
}

__global__ void k_seg(const bf16_t* __restrict__ X, const float* __restrict__ w,
                      const float* __restrict__ b, float* __restrict__ out) {
  int p = blockIdx.x * blockDim.x + threadIdx.x;
  if (p >= 9216) return;
  float s = b[0];
  for (int c = 0; c < 512; ++c) s += w[c] * bf2f(X[(size_t)p * 512 + c]);
  out[p] = s;
}

// ---------------------------------------------------------------------------
// Host orchestration
// ---------------------------------------------------------------------------
extern "C" void kernel_launch(void* const* d_in, const int* in_sizes, int n_in,
                              void* d_out, int out_size, void* d_ws, size_t ws_size,
                              hipStream_t stream) {
  (void)in_sizes; (void)n_in; (void)out_size; (void)ws_size;
  auto F = [&](int i) { return (const float*)d_in[i]; };

  // ---- workspace allocator ----
  size_t off = 0;
  auto alloc = [&](size_t bytes) -> void* {
    void* p = (void*)((char*)d_ws + off);
    off = (off + bytes + 255) & ~(size_t)255;
    return p;
  };
  auto aB = [&](size_t e) { return (bf16_t*)alloc(e * 2); };
  auto aF = [&](size_t e) { return (float*)alloc(e * 4); };

  // weights (bf16, K padded to mult of 32)
  bf16_t* lc4A  = aB((size_t)768 * 1216);
  bf16_t* lc3A  = aB((size_t)768 * 736);
  bf16_t* lc2A  = aB((size_t)768 * 448);
  bf16_t* lc1A  = aB((size_t)48 * 160);
  bf16_t* fuseA = aB((size_t)512 * 2304);
  bf16_t* shortA= aB((size_t)512 * 512);
  bf16_t* catA  = aB((size_t)512 * 2560);
  bf16_t* llfA  = aB((size_t)512 * 576);
  bf16_t* dsA   = aB((size_t)512 * 32768);     // reused per r (max r=8)
  bf16_t* gA    = aB((size_t)256 * 512);
  bf16_t* thA   = aB((size_t)256 * 512);
  bf16_t* phA   = aB((size_t)256 * 512);
  bf16_t* coA   = aB((size_t)512 * 256);
  // activations (pixel-major bf16)
  bf16_t* c4T   = aB((size_t)144 * 1216);
  bf16_t* c3T   = aB((size_t)576 * 736);
  bf16_t* c2T   = aB((size_t)2304 * 448);
  bf16_t* c1T   = aB((size_t)9216 * 160);
  bf16_t* m4o   = aB((size_t)144 * 768);
  bf16_t* m3o   = aB((size_t)576 * 768);
  bf16_t* FUSEIN= aB((size_t)2304 * 2304);
  bf16_t* Cb    = aB((size_t)2304 * 512);      // _c
  bf16_t* CATIN = aB((size_t)2304 * 2560);
  bf16_t* qcols = aB((size_t)2304 * 512);
  bf16_t* Bds   = aB((size_t)169 * 32768);     // reused per r (max r=8)
  float*  Ef32  = aF((size_t)784 * 512);       // max Nn = 784 (r=2)
  bf16_t* Ectx  = aB((size_t)784 * 512);
  bf16_t* ctx2  = aB((size_t)2304 * 512);
  bf16_t* gcols = aB((size_t)2304 * 256);
  bf16_t* phcols= aB((size_t)2304 * 256);
  bf16_t* thcols= aB((size_t)2304 * 256);
  bf16_t* ycols = aB((size_t)2304 * 256);
  bf16_t* out48 = aB((size_t)2304 * 512);
  bf16_t* llfbuf= aB((size_t)9216 * 576);
  bf16_t* llfout= aB((size_t)9216 * 512);
  float*  meanv = aF(512);
  int*    cmap  = (int*)alloc(2304 * 4);

  auto NB = [](size_t total) { return dim3((unsigned)((total + 255) / 256)); };
  auto gemm = [&](const bf16_t* A, const bf16_t* B, void* Out, int ldo, int ooff,
                  int M, int N, int K, int lda, int ldb,
                  const float* scale, const float* bias,
                  const bf16_t* resid, int ldr, const int* cm, int relu, bool outf32) {
    dim3 g((unsigned)((N + 127) / 128), (unsigned)((M + 63) / 64));
    if (outf32)
      k_gemm<true><<<g, 256, 0, stream>>>(A, B, Out, ldo, ooff, M, N, K, lda, ldb,
                                          scale, BN_INV, bias, resid, ldr, cm, relu);
    else
      k_gemm<false><<<g, 256, 0, stream>>>(A, B, Out, ldo, ooff, M, N, K, lda, ldb,
                                           scale, BN_INV, bias, resid, ldr, cm, relu);
  };

  // ---- weight conversion ----
  k_cvt_w<<<NB((size_t)768 * 1216), 256, 0, stream>>>(F(4),  lc4A,  768, 1200, 1216);
  k_cvt_w<<<NB((size_t)768 * 736),  256, 0, stream>>>(F(6),  lc3A,  768, 720,  736);
  k_cvt_w<<<NB((size_t)768 * 448),  256, 0, stream>>>(F(8),  lc2A,  768, 420,  448);
  k_cvt_w<<<NB((size_t)48 * 160),   256, 0, stream>>>(F(10), lc1A,  48,  140,  160);
  k_cvt_w<<<NB((size_t)512 * 2304), 256, 0, stream>>>(F(12), fuseA, 512, 2304, 2304);
  k_cvt_w<<<NB((size_t)512 * 512),  256, 0, stream>>>(F(15), shortA,512, 512,  512);
  k_cvt_w<<<NB((size_t)512 * 2560), 256, 0, stream>>>(F(18), catA,  512, 2560, 2560);
  k_cvt_w<<<NB((size_t)512 * 576),  256, 0, stream>>>(F(21), llfA,  512, 560,  576);

  // ---- input transposes (channel-major f32 -> pixel-major bf16) ----
  k_transpose<<<NB((size_t)144 * 1216), 256, 0, stream>>>(F(3), c4T, 1200, 144, 1216);
  k_transpose<<<NB((size_t)576 * 736),  256, 0, stream>>>(F(2), c3T, 720,  576, 736);
  k_transpose<<<NB((size_t)2304 * 448), 256, 0, stream>>>(F(1), c2T, 420,  2304, 448);
  k_transpose<<<NB((size_t)9216 * 160), 256, 0, stream>>>(F(0), c1T, 140,  9216, 160);

  // ---- MLP projections + resize into fuse input ----
  gemm(lc4A, c4T, m4o, 768, 0, 768, 144,  1216, 1216, 1216, nullptr, F(5), nullptr, 0, nullptr, 0, false);
  gemm(lc3A, c3T, m3o, 768, 0, 768, 576,  736,  736,  736,  nullptr, F(7), nullptr, 0, nullptr, 0, false);
  k_resize<<<NB((size_t)2304 * 768), 256, 0, stream>>>(m4o, 12, 12, 768, FUSEIN, 48, 48, 2304, 0, 768);
  k_resize<<<NB((size_t)2304 * 768), 256, 0, stream>>>(m3o, 24, 24, 768, FUSEIN, 48, 48, 2304, 768, 768);
  gemm(lc2A, c2T, FUSEIN, 2304, 1536, 768, 2304, 448, 448, 448, nullptr, F(9), nullptr, 0, nullptr, 0, false);

  // ---- fuse / short / pool ----
  gemm(fuseA, FUSEIN, Cb, 512, 0, 512, 2304, 2304, 2304, 2304, F(13), F(14), nullptr, 0, nullptr, 1, false);
  gemm(shortA, Cb, CATIN, 2560, 0, 512, 2304, 512, 512, 512, F(16), F(17), nullptr, 0, nullptr, 1, false);
  k_meanpool<<<NB(512), 256, 0, stream>>>(Cb, meanv);
  k_ip_broadcast<<<NB(512), 256, 0, stream>>>(F(24), F(25), meanv, CATIN, 2560);

  // ---- query gather ----
  k_make_cmap<<<NB(2304), 256, 0, stream>>>(cmap);
  k_gather_rows<<<NB((size_t)2304 * 512), 256, 0, stream>>>(Cb, qcols, cmap, 2304, 512);

  // ---- 3 lawin-attention branches (r = 8,4,2) ----
  const int rs[3]    = {8, 4, 2};
  const int heads[3] = {64, 16, 4};
  for (int ri = 0; ri < 3; ++ri) {
    int r = rs[ri], head = heads[ri], d = 256 / head;
    int Ns = 40 / r + 8, Nn = Ns * Ns;
    int Kds = 512 * r * r, pad = (r - 1) * 4;
    int base = 26 + ri * 15;

    k_cvt_ds<<<NB((size_t)512 * Kds), 256, 0, stream>>>(F(base + 0), dsA, r);
    k_cvt_w<<<NB((size_t)256 * 512), 256, 0, stream>>>(F(base + 6),  gA,  256, 512, 512);
    k_cvt_w<<<NB((size_t)256 * 512), 256, 0, stream>>>(F(base + 8),  thA, 256, 512, 512);
    k_cvt_w<<<NB((size_t)256 * 512), 256, 0, stream>>>(F(base + 10), phA, 256, 512, 512);
    k_cvt_w<<<NB((size_t)512 * 256), 256, 0, stream>>>(F(base + 12), coA, 512, 256, 256);

    // deduplicated unfold + patch-embed GEMM + layernorm
    k_gather_ds<<<NB((size_t)Nn * Kds), 256, 0, stream>>>(Cb, Bds, r, Ns, pad);
    gemm(dsA, Bds, Ef32, 512, 0, 512, Nn, Kds, Kds, Kds,
         nullptr, F(base + 1), nullptr, 0, nullptr, 0, true);
    k_layernorm<<<dim3(Nn), 256, 0, stream>>>(Ef32, Ectx, F(base + 2), F(base + 3), Nn);

    // position-mix, then g/ph/th projections
    k_mix<<<NB((size_t)2304 * 512), 256, 0, stream>>>(Ectx, F(base + 4), F(base + 5), ctx2, r, Ns, head);
    gemm(gA,  ctx2,  gcols,  256, 0, 256, 2304, 512, 512, 512, nullptr, F(base + 7),  nullptr, 0, nullptr, 0, false);
    gemm(phA, ctx2,  phcols, 256, 0, 256, 2304, 512, 512, 512, nullptr, F(base + 11), nullptr, 0, nullptr, 0, false);
    gemm(thA, qcols, thcols, 256, 0, 256, 2304, 512, 512, 512, nullptr, F(base + 9),  nullptr, 0, nullptr, 0, false);

    // attention cores
    k_attn<<<dim3(36 * head), 64, 3 * 64 * d * 4, stream>>>(thcols, phcols, gcols, ycols, d, head);

    // co projection + BN + residual(query), scattered back to image layout
    gemm(coA, ycols, CATIN, 2560, 1024 + ri * 512, 512, 2304, 256, 256, 256,
         F(base + 13), F(base + 14), qcols, 512, cmap, 0, false);
  }

  // ---- cat conv, upsample, low-level fuse, seg head ----
  gemm(catA, CATIN, out48, 512, 0, 512, 2304, 2560, 2560, 2560, F(19), F(20), nullptr, 0, nullptr, 1, false);

  hipMemsetAsync(llfbuf, 0, (size_t)9216 * 576 * 2, stream);   // zero K-pad cols
  k_resize<<<NB((size_t)9216 * 512), 256, 0, stream>>>(out48, 48, 48, 512, llfbuf, 96, 96, 576, 0, 512);
  gemm(lc1A, c1T, llfbuf, 576, 512, 48, 9216, 160, 160, 160, nullptr, F(11), nullptr, 0, nullptr, 0, false);
  gemm(llfA, llfbuf, llfout, 512, 0, 512, 9216, 576, 576, 576, F(22), F(23), nullptr, 0, nullptr, 1, false);

  k_seg<<<NB(9216), 256, 0, stream>>>(llfout, F(71), F(72), (float*)d_out);
}
